// TemporalToSpatialAttention_51436528336965
// MI455X (gfx1250) — matrix operand order repaired and checked by reading, not compiled, not run on hardware
//
#include <hip/hip_runtime.h>
#include <hip/hip_bf16.h>

typedef __attribute__((ext_vector_type(16))) __bf16 v16bf;
typedef __attribute__((ext_vector_type(8)))  __bf16 v8bf;
typedef __attribute__((ext_vector_type(8)))  float  v8f;
typedef __attribute__((ext_vector_type(4)))  unsigned short us4;

#define HIDDEN   256
#define HEADS    8
#define HEAD_DIM 32
#define LTEMP    24
#define ROWS     4096   // B*N = 8*512

// ---------------- helpers ----------------

__device__ __forceinline__ unsigned short f2bf(float f) {
  unsigned u = __float_as_uint(f);
  u += 0x7FFFu + ((u >> 16) & 1u);      // round-to-nearest-even
  return (unsigned short)(u >> 16);
}
__device__ __forceinline__ float bf2f(unsigned short h) {
  return __uint_as_float(((unsigned)h) << 16);
}
__device__ __forceinline__ us4 f2bf4(float4 v) {
  us4 r;
  r.x = f2bf(v.x); r.y = f2bf(v.y); r.z = f2bf(v.z); r.w = f2bf(v.w);
  return r;
}

// Build a 16-element bf16 WMMA fragment from two contiguous 16B LDS loads.
// A-fragment (16x32): hiOff = 16 (VGPR4-7 hold K+16..K+23 per ISA layout)
// B-fragment (32x16): hiOff = 16 (same K order as A-fragment)
__device__ __forceinline__ v16bf ld_frag(const unsigned short* p, int hiOff) {
  v8bf lo = *(const v8bf*)p;
  v8bf hi = *(const v8bf*)(p + hiOff);
  v16bf r;
#pragma unroll
  for (int i = 0; i < 8; ++i) { r[i] = lo[i]; r[8 + i] = hi[i]; }
  return r;
}

// ---------------- generic GEMM: C[M,256] = A[M,256] @ B[256,256] + bias ----------------

#define G_LDK 264                 // 256 + 8 bf16 pad
#define G_MT  32                  // rows per workgroup
#define GEMM_LDS ((G_MT + 256) * G_LDK * 2 + 1024)

__global__ __launch_bounds__(256)
void gemm256_bias_kernel(const float* __restrict__ A, const float* __restrict__ B,
                         const float* __restrict__ bias, float* __restrict__ C) {
  extern __shared__ char smem[];
  unsigned short* sA   = (unsigned short*)smem;                 // [32][264] bf16
  unsigned short* sBT  = sA + G_MT * G_LDK;                     // [256][264] bf16 (B^T)
  float*          sBia = (float*)(smem + (G_MT + 256) * G_LDK * 2);
  const int tid = threadIdx.x;
  const int rowBase = blockIdx.x * G_MT;

  // B[k][n] -> BT[n][k] bf16, float4 global loads
  const float4* B4 = (const float4*)B;
  for (int i4 = tid; i4 < 256 * 256 / 4; i4 += 256) {
    float4 v = B4[i4];
    int k = i4 >> 6, n = (i4 << 2) & 255;
    sBT[(n + 0) * G_LDK + k] = f2bf(v.x);
    sBT[(n + 1) * G_LDK + k] = f2bf(v.y);
    sBT[(n + 2) * G_LDK + k] = f2bf(v.z);
    sBT[(n + 3) * G_LDK + k] = f2bf(v.w);
  }
  sBia[tid] = bias[tid];
  // A tile -> bf16, float4 loads, b64 LDS stores
  const float4* A4 = (const float4*)(A + (size_t)rowBase * 256);
  for (int i4 = tid; i4 < G_MT * 256 / 4; i4 += 256) {
    float4 v = A4[i4];
    int r = i4 >> 6, col = (i4 << 2) & 255;
    *(us4*)&sA[r * G_LDK + col] = f2bf4(v);
  }
  __syncthreads();

  const int wave = tid >> 5, lane = tid & 31;
  const int lm = lane & 15, lk = lane >> 4;
  const int mt = wave >> 2;             // 0..1  (M tile)
  const int ntBase = (wave & 3) * 4;    // N tiles ntBase..ntBase+3

  v8f acc[4] = {};
  for (int ks = 0; ks < 8; ++ks) {
    const int k0 = ks * 32;
    v16bf a = ld_frag(&sA[(mt * 16 + lm) * G_LDK + k0 + lk * 8], 16);
#pragma unroll
    for (int j = 0; j < 4; ++j) {
      v16bf b = ld_frag(&sBT[((ntBase + j) * 16 + lm) * G_LDK + k0 + lk * 8], 16);
      acc[j] = __builtin_amdgcn_wmma_f32_16x16x32_bf16(
          false, a, false, b, (short)0, acc[j], false, false);
    }
  }

#pragma unroll
  for (int j = 0; j < 4; ++j) {
    const int col = (ntBase + j) * 16 + lm;
    const float bi = sBia[col];
#pragma unroll
    for (int r = 0; r < 8; ++r) {
      const int row = rowBase + mt * 16 + lk * 8 + r;
      C[(size_t)row * 256 + col] = acc[j][r] + bi;
    }
  }
}

// ---------------- fused kv-GEMM + per-(b,n) attention ----------------

#define A_LDK 264
#define KV_LD 520
#define CHUNK 8
#define ATTN_SHORTS (512 * A_LDK + 32 * A_LDK + LTEMP * KV_LD)   // 156096
#define ATTN_QOFF   (ATTN_SHORTS * 2)                            // 312192 B
#define ATTN_BOFF   (ATTN_QOFF + 256 * 4)                        // 313216 B
#define ATTN_LDS    (ATTN_BOFF + 512 * 4)                        // 315264 B

__global__ __launch_bounds__(256)
void attn_kernel(const float* __restrict__ Xt,    // [4096][24][256] temporal
                 const float* __restrict__ Wkv,   // [256][512]
                 const float* __restrict__ bkv,   // [512]
                 const float* __restrict__ Qws,   // [4096][256] q = spatial@Wq+bq
                 float* __restrict__ Ows) {       // [4096][256] attention output
  extern __shared__ char smem[];
  unsigned short* sWB = (unsigned short*)smem;       // WkvT [512][264] bf16 (resident)
  unsigned short* sA  = sWB + 512 * A_LDK;           // [32][264] bf16 (rows 24..31 zero)
  unsigned short* sKV = sA + 32 * A_LDK;             // [24][520] bf16 (k | v)
  float* sQ   = (float*)(smem + ATTN_QOFF);          // [256]
  float* sBkv = (float*)(smem + ATTN_BOFF);          // [512]
  const int tid = threadIdx.x;

  // One-time: Wkv[k][n] -> WkvT[n][k] bf16 (resident for all CHUNK iterations)
  const float4* W4 = (const float4*)Wkv;
  for (int i4 = tid; i4 < 256 * 512 / 4; i4 += 256) {
    float4 v = W4[i4];
    int k = i4 >> 7, n = (i4 << 2) & 511;
    sWB[(n + 0) * A_LDK + k] = f2bf(v.x);
    sWB[(n + 1) * A_LDK + k] = f2bf(v.y);
    sWB[(n + 2) * A_LDK + k] = f2bf(v.z);
    sWB[(n + 3) * A_LDK + k] = f2bf(v.w);
  }
  for (int i = tid; i < 512; i += 256) sBkv[i] = bkv[i];
  for (int i = tid; i < 8 * A_LDK; i += 256) sA[24 * A_LDK + i] = 0;  // zero M-pad rows

  const int wave = tid >> 5, lane = tid & 31;
  const int lm = lane & 15, lk = lane >> 4;

  // Register double-buffer: preload iteration 0 tile (6 x float4 / thread) + q row
  const int flat0 = blockIdx.x * CHUNK;
  float4 rb[6];
  {
    const float4* xt4 = (const float4*)(Xt + (size_t)flat0 * (LTEMP * 256));
#pragma unroll
    for (int i = 0; i < 6; ++i) rb[i] = xt4[tid + i * 256];
  }
  float qr = Qws[(size_t)flat0 * 256 + tid];
  __syncthreads();

  for (int it = 0; it < CHUNK; ++it) {
    const int flat = flat0 + it;

    // stage current tile into LDS as bf16 (b64 stores)
#pragma unroll
    for (int i = 0; i < 6; ++i) {
      int i4 = tid + i * 256;
      int r = i4 >> 6, col = (i4 << 2) & 255;
      *(us4*)&sA[r * A_LDK + col] = f2bf4(rb[i]);
    }
    sQ[tid] = qr;
    __syncthreads();

    // issue next iteration's global loads now; they complete during WMMA+attention
    if (it + 1 < CHUNK) {
      const float4* xt4 = (const float4*)(Xt + (size_t)(flat + 1) * (LTEMP * 256));
#pragma unroll
      for (int i = 0; i < 6; ++i) rb[i] = xt4[tid + i * 256];
      qr = Qws[(size_t)(flat + 1) * 256 + tid];
    }

    // kv(32x512) = A(32x256) @ Wkv(256x512); wave w owns N-tiles 4w..4w+3, both M-tiles
    v8f acc[2][4] = {};
    for (int ks = 0; ks < 8; ++ks) {
      const int k0 = ks * 32;
      v16bf a0 = ld_frag(&sA[lm * A_LDK + k0 + lk * 8], 16);
      v16bf a1 = ld_frag(&sA[(16 + lm) * A_LDK + k0 + lk * 8], 16);
#pragma unroll
      for (int j = 0; j < 4; ++j) {
        const int n = (wave * 4 + j) * 16 + lm;
        v16bf b = ld_frag(&sWB[n * A_LDK + k0 + lk * 8], 16);
        acc[0][j] = __builtin_amdgcn_wmma_f32_16x16x32_bf16(
            false, a0, false, b, (short)0, acc[0][j], false, false);
        acc[1][j] = __builtin_amdgcn_wmma_f32_16x16x32_bf16(
            false, a1, false, b, (short)0, acc[1][j], false, false);
      }
    }

    // kv + bias -> LDS bf16 (only real rows m<24)
#pragma unroll
    for (int mtI = 0; mtI < 2; ++mtI) {
#pragma unroll
      for (int j = 0; j < 4; ++j) {
        const int nn = (wave * 4 + j) * 16 + lm;
        const float bi = sBkv[nn];
#pragma unroll
        for (int r = 0; r < 8; ++r) {
          const int m = mtI * 16 + lk * 8 + r;
          if (m < LTEMP) sKV[m * KV_LD + nn] = f2bf(acc[mtI][j][r] + bi);
        }
      }
    }
    __syncthreads();

    // attention: wave == head h, lane == d; k = sKV[l][h*32+d], v = sKV[l][256+h*32+d]
    const int h = wave, d = lane;
    const float qv = sQ[h * 32 + d];
    float s[LTEMP];
#pragma unroll
    for (int l = 0; l < LTEMP; ++l) {
      float p = qv * bf2f(sKV[l * KV_LD + h * 32 + d]);
#pragma unroll
      for (int off = 16; off > 0; off >>= 1) p += __shfl_xor(p, off, 32);
      s[l] = p * 0.17677669529663687f;            // 1/sqrt(32)
    }
    float mx = s[0];
#pragma unroll
    for (int l = 1; l < LTEMP; ++l) mx = fmaxf(mx, s[l]);
    float sum = 0.f;
#pragma unroll
    for (int l = 0; l < LTEMP; ++l) { s[l] = __expf(s[l] - mx); sum += s[l]; }
    const float inv = 1.f / sum;
    float o = 0.f;
#pragma unroll
    for (int l = 0; l < LTEMP; ++l)
      o += s[l] * bf2f(sKV[l * KV_LD + 256 + h * 32 + d]);
    Ows[(size_t)flat * 256 + h * 32 + d] = o * inv;
    __syncthreads();
  }
}

// ---------------- launch ----------------

extern "C" void kernel_launch(void* const* d_in, const int* in_sizes, int n_in,
                              void* d_out, int out_size, void* d_ws, size_t ws_size,
                              hipStream_t stream) {
  const float* spatial  = (const float*)d_in[0];
  const float* temporal = (const float*)d_in[1];
  const float* Wq  = (const float*)d_in[2];
  const float* bq  = (const float*)d_in[3];
  const float* Wkv = (const float*)d_in[4];
  const float* bkv = (const float*)d_in[5];
  const float* Wo  = (const float*)d_in[6];
  const float* bo  = (const float*)d_in[7];
  float* out     = (float*)d_out;
  float* q_ws    = (float*)d_ws;                        // [4096][256]
  float* attn_ws = q_ws + (size_t)ROWS * 256;           // [4096][256]

  hipFuncSetAttribute((const void*)gemm256_bias_kernel,
                      hipFuncAttributeMaxDynamicSharedMemorySize, GEMM_LDS);
  hipFuncSetAttribute((const void*)attn_kernel,
                      hipFuncAttributeMaxDynamicSharedMemorySize, ATTN_LDS);

  gemm256_bias_kernel<<<ROWS / G_MT, 256, GEMM_LDS, stream>>>(spatial, Wq, bq, q_ws);
  attn_kernel<<<ROWS / CHUNK, 256, ATTN_LDS, stream>>>(temporal, Wkv, bkv, q_ws, attn_ws);
  gemm256_bias_kernel<<<ROWS / G_MT, 256, GEMM_LDS, stream>>>(attn_ws, Wo, bo, out);
}
